// FastSpeech2GPU_52596169506894
// MI455X (gfx1250) — matrix operand order repaired and checked by reading, not compiled, not run on hardware
//
#include <hip/hip_runtime.h>
#include <hip/hip_bf16.h>

// ---------------------------------------------------------------------------
// FastSpeech2 forward on gfx1250. All GEMMs + attention use
// v_wmma_f32_16x16x32_bf16 (wave32 WMMA, fp32 accumulate).
// LDS tiles are stored in *fragment-native* layout [group][lane][j] so each
// lane's v16bf fragment is 32 contiguous bytes (2x ds_load_b128), and staging
// uses float4 global loads + 16B ds_store_b128 per 8-element octet.
// ---------------------------------------------------------------------------

typedef __attribute__((ext_vector_type(16))) __bf16 v16bf;
typedef __attribute__((ext_vector_type(8)))  __bf16 v8bf;
typedef __attribute__((ext_vector_type(8)))  float  v8f;

#define WMMA_BF16(a, b, c) \
  __builtin_amdgcn_wmma_f32_16x16x32_bf16(false, (a), false, (b), (short)0, (c), false, false)

#define B_   32
#define T_   256
#define D_   256
#define FF_  1024
#define NM_  80
#define NH_  4
#define DH_  64
#define MB_  1024   // MAX_BUF

__device__ __forceinline__ float gelu_exact(float x) {
  return 0.5f * x * (1.0f + erff(x * 0.70710678118654752f));
}

__device__ __forceinline__ v8bf pack8(const float4 f0, const float4 f1) {
  v8bf p;
  p[0] = (__bf16)f0.x; p[1] = (__bf16)f0.y; p[2] = (__bf16)f0.z; p[3] = (__bf16)f0.w;
  p[4] = (__bf16)f1.x; p[5] = (__bf16)f1.y; p[6] = (__bf16)f1.z; p[7] = (__bf16)f1.w;
  return p;
}

// ---------------------------------------------------------------------------
// Generic GEMM:  Y[M,N] = act( X[M,K] @ W[N,K]^T + bias[N] )
// act: 0 = none, 1 = exact GELU.  M % 128 == 0, K % 32 == 0, any N (guarded).
// Block: 256 threads (8 waves), tile 128(M) x 64(N), K-step 32.
//
// Fragment-native LDS layouts:
//  A (16x32 tile, A-packing): lane = (m&15) + 16*((k>>3)&1),
//                             j    = (k&7)  +  8*((k>>4)&1)
//  B (32x16 tile, B-packing): lane = (n&15) + 16*((k>>3)&1), j = (k&7) + 8*((k>>4)&1)
// ---------------------------------------------------------------------------
__global__ __launch_bounds__(256)
void k_gemm(const float* __restrict__ X, const float* __restrict__ W,
            const float* __restrict__ bias, float* __restrict__ Y,
            int M, int N, int K, int act)
{
  __shared__ alignas(16) __bf16 As2[8][32][16];   // 8 m-tiles of 16 rows
  __shared__ alignas(16) __bf16 Bs2[4][32][16];   // 4 n-groups of 16 cols

  const int tid  = threadIdx.x;
  const int lane = tid & 31;
  const int wave = tid >> 5;
  const int wm = (wave & 3) * 32;   // 0,32,64,96
  const int wn = (wave >> 2) * 32;  // 0,32
  const int bm = blockIdx.y * 128;
  const int bn = blockIdx.x * 64;
  const int arow = lane & 15;
  const int roff = (lane >> 4) << 3;

  v8f acc00 = {}, acc01 = {}, acc10 = {}, acc11 = {};

  for (int k0 = 0; k0 < K; k0 += 32) {
    // ---- stage A (128x32): 512 octets, 2 per thread, vectorized ----
#pragma unroll
    for (int i = 0; i < 2; ++i) {
      int idx = tid + i * 256;                 // 0..511
      int row = idx >> 2, oct = idx & 3;       // k-octet
      const float* src = &X[(size_t)(bm + row) * K + (k0 + oct * 8)];
      float4 f0 = *(const float4*)src;
      float4 f1 = *(const float4*)(src + 4);
      int ln = (row & 15) + ((oct & 1) << 4);
      int j0 = (oct >> 1) << 3;
      *(v8bf*)&As2[row >> 4][ln][j0] = pack8(f0, f1);
    }
    // ---- stage B = W^T (32x64): 256 octets, 1 per thread ----
    {
      int col = tid >> 2, oct = tid & 3;
      int gcol = bn + col;
      v8bf p = {};
      if (gcol < N) {
        const float* src = &W[(size_t)gcol * K + (k0 + oct * 8)];
        float4 f0 = *(const float4*)src;
        float4 f1 = *(const float4*)(src + 4);
        p = pack8(f0, f1);
      }
      int ln = (col & 15) + ((oct & 1) << 4);
      int j0 = (oct >> 1) << 3;
      *(v8bf*)&Bs2[col >> 4][ln][j0] = p;
    }
    if (k0 + 32 < K) {
      __builtin_prefetch(&X[(size_t)(bm + (tid >> 1)) * K + (k0 + 32)], 0, 1);
    }
    __syncthreads();

    // ---- fragments: contiguous 32B per lane -> 2x ds_load_b128 each ----
    v16bf a0 = *(const v16bf*)&As2[(wm >> 4)][lane][0];
    v16bf a1 = *(const v16bf*)&As2[(wm >> 4) + 1][lane][0];
    v16bf b0 = *(const v16bf*)&Bs2[(wn >> 4)][lane][0];
    v16bf b1 = *(const v16bf*)&Bs2[(wn >> 4) + 1][lane][0];

    acc00 = WMMA_BF16(a0, b0, acc00);
    acc01 = WMMA_BF16(a0, b1, acc01);
    acc10 = WMMA_BF16(a1, b0, acc10);
    acc11 = WMMA_BF16(a1, b1, acc11);
    __syncthreads();
  }

  // ---- epilogue: bias + activation + guarded store ----
  const int colA = bn + wn + arow;
  const int colB = bn + wn + 16 + arow;
  const float bA = (colA < N) ? bias[colA] : 0.0f;
  const float bB = (colB < N) ? bias[colB] : 0.0f;
#pragma unroll
  for (int r = 0; r < 8; ++r) {
    int rowA = bm + wm + r + roff;
    int rowB = bm + wm + 16 + r + roff;
    float v00 = acc00[r] + bA, v01 = acc01[r] + bB;
    float v10 = acc10[r] + bA, v11 = acc11[r] + bB;
    if (act == 1) {
      v00 = gelu_exact(v00); v01 = gelu_exact(v01);
      v10 = gelu_exact(v10); v11 = gelu_exact(v11);
    }
    if (colA < N) { Y[(size_t)rowA * N + colA] = v00; Y[(size_t)rowB * N + colA] = v10; }
    if (colB < N) { Y[(size_t)rowA * N + colB] = v01; Y[(size_t)rowB * N + colB] = v11; }
  }
}

// ---------------------------------------------------------------------------
// Flash attention: 128 threads = 4 waves, each wave owns a 16-query tile
// (block covers 64 queries).  K/V chunk staging (32 keys) is cooperative and
// vectorized; fragments load as contiguous v16bf from fragment-native LDS.
// ---------------------------------------------------------------------------
__global__ __launch_bounds__(128)
void k_attn(const float* __restrict__ Q, const float* __restrict__ K,
            const float* __restrict__ V, float* __restrict__ O,
            const int* __restrict__ lens, int use_mask, int Tq, int Tk,
            long long qbat, int qrow, long long kbat, int krow,
            long long vbat, int vrow)
{
  const int tid  = threadIdx.x;
  const int lane = tid & 31;
  const int w    = tid >> 5;
  const int r0 = blockIdx.x * 64 + w * 16;
  const int h  = blockIdx.y;
  const int b  = blockIdx.z;

  const float* q = Q + (long long)b * qbat + h * DH_;
  const float* k = K + (long long)b * kbat + h * DH_;
  const float* v = V + (long long)b * vbat + h * DH_;
  const int len = use_mask ? lens[b] : Tk;

  __shared__ alignas(16) __bf16 Qs[4][2][32][16];   // per-wave Q A-fragments
  __shared__ alignas(16) __bf16 Ks[2][2][32][16];   // [d-half][key-grp][lane][j]
  __shared__ alignas(16) __bf16 Vs[4][32][16];      // [d-grp][lane][j]
  __shared__ float  Ss[4][16][32];
  __shared__ alignas(16) __bf16 Ps[4][32][16];
  __shared__ float  mrow[4][16], lrow[4][16], srow[4][16];

  const int arow = lane & 15;
  const int roff = (lane >> 4) << 3;

  // ---- stage Q (once): 128 octets per wave, 4 per lane ----
#pragma unroll
  for (int i = 0; i < 4; ++i) {
    int idx = lane + i * 32;                 // 0..127
    int row = idx >> 3, oct = idx & 7;       // oct: 8-float octet over d 0..63
    const float* src = &q[(r0 + row) * qrow + oct * 8];
    float4 f0 = *(const float4*)src;
    float4 f1 = *(const float4*)(src + 4);
    int f  = oct >> 2;
    int o2 = oct & 3;
    int ln = (row & 15) + ((o2 & 1) << 4);
    int j0 = (o2 >> 1) << 3;
    *(v8bf*)&Qs[w][f][ln][j0] = pack8(f0, f1);
  }
  if (lane < 16) { mrow[w][lane] = -3.0e38f; lrow[w][lane] = 0.0f; }
  __syncthreads();
  v16bf aq0 = *(const v16bf*)&Qs[w][0][lane][0];
  v16bf aq1 = *(const v16bf*)&Qs[w][1][lane][0];
  v8f acc0 = {}, acc1 = {}, acc2 = {}, acc3 = {};

  for (int kc = 0; kc < Tk; kc += 32) {
    // ---- stage K chunk (32 keys x 64 d) as S-GEMM B-fragments ----
#pragma unroll
    for (int i = 0; i < 2; ++i) {
      int idx = tid + i * 128;               // 0..255
      int key = idx >> 3, oct = idx & 7;
      const float* src = &k[(kc + key) * krow + oct * 8];
      float4 f0 = *(const float4*)src;
      float4 f1 = *(const float4*)(src + 4);
      int f  = oct >> 2;
      int o2 = oct & 3;
      int ln = (key & 15) + ((o2 & 1) << 4);
      int j0 = (o2 >> 1) << 3;
      *(v8bf*)&Ks[f][key >> 4][ln][j0] = pack8(f0, f1);
    }
    // ---- stage V chunk (32 keys x 64 d) as PV B-fragments (transpose) ----
#pragma unroll
    for (int i = 0; i < 2; ++i) {
      int idx = tid + i * 128;               // 0..255
      int d = idx >> 2, ko = idx & 3;        // ko: key-octet
      v8bf p;
#pragma unroll
      for (int j = 0; j < 8; ++j) p[j] = (__bf16)v[(kc + ko * 8 + j) * vrow + d];
      int ln = (d & 15) + ((ko & 1) << 4);
      int j0 = (ko >> 1) << 3;
      *(v8bf*)&Vs[d >> 4][ln][j0] = p;
    }
    __syncthreads();

    // ---- S = Q K^T (two 16-key halves) ----
    v8f s0 = {}, s1 = {};
    {
      v16bf b00 = *(const v16bf*)&Ks[0][0][lane][0];
      v16bf b10 = *(const v16bf*)&Ks[1][0][lane][0];
      v16bf b01 = *(const v16bf*)&Ks[0][1][lane][0];
      v16bf b11 = *(const v16bf*)&Ks[1][1][lane][0];
      s0 = WMMA_BF16(aq0, b00, s0);
      s0 = WMMA_BF16(aq1, b10, s0);
      s1 = WMMA_BF16(aq0, b01, s1);
      s1 = WMMA_BF16(aq1, b11, s1);
    }
    // scale 1/sqrt(64) + key-padding mask, spill per-wave S tile
    const int col0 = kc + arow, col1 = kc + 16 + arow;
    const bool m0 = use_mask && (col0 >= len);
    const bool m1 = use_mask && (col1 >= len);
#pragma unroll
    for (int r = 0; r < 8; ++r) {
      Ss[w][r + roff][arow]      = m0 ? -1.0e9f : s0[r] * 0.125f;
      Ss[w][r + roff][16 + arow] = m1 ? -1.0e9f : s1[r] * 0.125f;
    }

    // ---- online softmax (wave-local; one row per lane 0..15) ----
    if (lane < 16) {
      const float4* row4 = (const float4*)&Ss[w][lane][0];
      float4 c0 = row4[0], c1 = row4[1], c2 = row4[2], c3 = row4[3];
      float4 c4 = row4[4], c5 = row4[5], c6 = row4[6], c7 = row4[7];
      float sv[32] = {c0.x,c0.y,c0.z,c0.w, c1.x,c1.y,c1.z,c1.w,
                      c2.x,c2.y,c2.z,c2.w, c3.x,c3.y,c3.z,c3.w,
                      c4.x,c4.y,c4.z,c4.w, c5.x,c5.y,c5.z,c5.w,
                      c6.x,c6.y,c6.z,c6.w, c7.x,c7.y,c7.z,c7.w};
      float mx = mrow[w][lane];
#pragma unroll
      for (int c = 0; c < 32; ++c) mx = fmaxf(mx, sv[c]);
      float sc = __expf(mrow[w][lane] - mx);
      float sm = 0.0f;
      float p[32];
#pragma unroll
      for (int c = 0; c < 32; ++c) { p[c] = __expf(sv[c] - mx); sm += p[c]; }
      lrow[w][lane] = lrow[w][lane] * sc + sm;
      mrow[w][lane] = mx;
      srow[w][lane] = sc;
      // write P in A-fragment order: lane m gets keys {0..7,16..23},
      // lane m+16 gets keys {8..15,24..31} -> two contiguous v16bf stores
      v16bf pa, pb;
#pragma unroll
      for (int j = 0; j < 8; ++j) {
        pa[j]     = (__bf16)p[j];
        pa[j + 8] = (__bf16)p[16 + j];
        pb[j]     = (__bf16)p[8 + j];
        pb[j + 8] = (__bf16)p[24 + j];
      }
      *(v16bf*)&Ps[w][lane][0]      = pa;
      *(v16bf*)&Ps[w][lane + 16][0] = pb;
    }

    // ---- rescale running accumulators ----
    float scl[8];
#pragma unroll
    for (int r = 0; r < 8; ++r) scl[r] = srow[w][r + roff];
#pragma unroll
    for (int r = 0; r < 8; ++r) {
      acc0[r] *= scl[r]; acc1[r] *= scl[r];
      acc2[r] *= scl[r]; acc3[r] *= scl[r];
    }

    // ---- O += P @ V ----
    v16bf ap = *(const v16bf*)&Ps[w][lane][0];
    v16bf bv0 = *(const v16bf*)&Vs[0][lane][0];
    v16bf bv1 = *(const v16bf*)&Vs[1][lane][0];
    v16bf bv2 = *(const v16bf*)&Vs[2][lane][0];
    v16bf bv3 = *(const v16bf*)&Vs[3][lane][0];
    acc0 = WMMA_BF16(ap, bv0, acc0);
    acc1 = WMMA_BF16(ap, bv1, acc1);
    acc2 = WMMA_BF16(ap, bv2, acc2);
    acc3 = WMMA_BF16(ap, bv3, acc3);
    __syncthreads();   // Ks/Vs consumed by all waves before next staging
  }

  // ---- normalize and store ----
  float linv[8];
#pragma unroll
  for (int r = 0; r < 8; ++r) {
    float L = lrow[w][r + roff];
    linv[r] = (L > 0.0f) ? (1.0f / L) : 0.0f;
  }
#pragma unroll
  for (int r = 0; r < 8; ++r) {
    long long row = r0 + r + roff;
    long long base = (long long)b * Tq * D_ + row * D_ + h * DH_ + arow;
    O[base +  0] = acc0[r] * linv[r];
    O[base + 16] = acc1[r] * linv[r];
    O[base + 32] = acc2[r] * linv[r];
    O[base + 48] = acc3[r] * linv[r];
  }
}

// ---------------------------------------------------------------------------
// LayerNorm over last dim (256), optional residual.  One row per block.
// ---------------------------------------------------------------------------
__global__ __launch_bounds__(256)
void k_ln(const float* __restrict__ X, const float* __restrict__ R,
          const float* __restrict__ w, const float* __restrict__ bb,
          float* __restrict__ Y, int has_res)
{
  __shared__ float red[256];
  const size_t row = blockIdx.x;
  const int t = threadIdx.x;
  float v = X[row * 256 + t];
  if (has_res) v += R[row * 256 + t];
  red[t] = v; __syncthreads();
  for (int s = 128; s > 0; s >>= 1) { if (t < s) red[t] += red[t + s]; __syncthreads(); }
  const float mean = red[0] * (1.0f / 256.0f);
  __syncthreads();
  red[t] = v * v; __syncthreads();
  for (int s = 128; s > 0; s >>= 1) { if (t < s) red[t] += red[t + s]; __syncthreads(); }
  const float var = red[0] * (1.0f / 256.0f) - mean * mean;
  const float inv = rsqrtf(var + 1e-5f);
  Y[row * 256 + t] = (v - mean) * inv * w[t] + bb[t];
}

// ---------------------------------------------------------------------------
// Conv1d k=3 pad=1 + ReLU on (B,256,256).  One (b, out-channel) per block.
// transposed=1: input is enc (B,T,D) read as (B,D,T); else channel-major.
// ---------------------------------------------------------------------------
__global__ __launch_bounds__(256)
void k_conv3(const float* __restrict__ X, const float* __restrict__ W,
             const float* __restrict__ Bv, float* __restrict__ Y, int transposed)
{
  __shared__ float wsm[768];
  const int o = blockIdx.x & 255;
  const int b = blockIdx.x >> 8;
  const int t = threadIdx.x;
  for (int i = t; i < 768; i += 256) wsm[i] = W[o * 768 + i];
  __syncthreads();
  const float* xb = X + (size_t)b * 65536;
  float s = Bv[o];
  for (int i = 0; i < 256; ++i) {
    const float w0 = wsm[i * 3], w1 = wsm[i * 3 + 1], w2 = wsm[i * 3 + 2];
    float xm = 0.0f, x0, xp = 0.0f;
    if (transposed) {
      if (t > 0)   xm = xb[(t - 1) * 256 + i];
      x0 = xb[t * 256 + i];
      if (t < 255) xp = xb[(t + 1) * 256 + i];
    } else {
      const float* xr = xb + i * 256;
      if (t > 0)   xm = xr[t - 1];
      x0 = xr[t];
      if (t < 255) xp = xr[t + 1];
    }
    s += w0 * xm + w1 * x0 + w2 * xp;
  }
  Y[(size_t)b * 65536 + o * 256 + t] = fmaxf(s, 0.0f);
}

// Final variance-predictor linear over last axis: out[b,c] = sum_t h[b,c,t]*lw[t]+lb
__global__ __launch_bounds__(256)
void k_vlin(const float* __restrict__ H, const float* __restrict__ lw,
            const float* __restrict__ lb, float* __restrict__ out)
{
  const int b = blockIdx.x, c = threadIdx.x;
  const float* h = H + (size_t)b * 65536 + c * 256;
  float s = lb[0];
  for (int tt = 0; tt < 256; ++tt) s += h[tt] * lw[tt];
  out[b * 256 + c] = s;
}

// ---------------------------------------------------------------------------
// Length regulator helpers
// ---------------------------------------------------------------------------
__global__ __launch_bounds__(256)
void k_rowsum(const float* __restrict__ dur, float* __restrict__ sums)
{
  __shared__ float red[256];
  const int b = blockIdx.x, t = threadIdx.x;
  red[t] = dur[b * 256 + t]; __syncthreads();
  for (int s = 128; s > 0; s >>= 1) { if (t < s) red[t] += red[t + s]; __syncthreads(); }
  if (t == 0) sums[b] = red[0];
}

__global__ void k_maxlen(const float* __restrict__ sums, int* __restrict__ ml)
{
  if (threadIdx.x == 0) {
    int mx = 0;
    for (int b = 0; b < B_; ++b) {
      int v = (int)rintf(sums[b]);
      if (v > mx) mx = v;
    }
    if (mx < 0) mx = 0;
    if (mx > MB_) mx = MB_;
    *ml = mx;
  }
}

__global__ void k_scan(const float* __restrict__ dur, const int* __restrict__ lens,
                       const int* __restrict__ maxlen, int* __restrict__ ends)
{
  const int b = blockIdx.x;
  const int ml = *maxlen;
  const int L = lens[b];
  int pos = 0;
  for (int t = 0; t < 256; ++t) {
    int d = (int)rintf(dur[b * 256 + t]);
    bool ok = (t < L) && (d > 0) && (pos + d <= ml);
    pos += ok ? d : 0;
    ends[b * 256 + t] = pos;
  }
}

__global__ __launch_bounds__(256)
void k_gather(const float* __restrict__ E, const int* __restrict__ ends,
              float* __restrict__ Y)
{
  const int f = blockIdx.x, b = blockIdx.y, d = threadIdx.x;
  const int* e = ends + b * 256;
  int lo = 0, hi = 256;
  while (lo < hi) { int mid = (lo + hi) >> 1; if (e[mid] <= f) lo = mid + 1; else hi = mid; }
  int tok = lo > 255 ? 255 : lo;
  float val = (f < e[255]) ? E[(size_t)b * 65536 + tok * 256 + d] : 0.0f;
  Y[(size_t)b * (MB_ * D_) + (size_t)f * D_ + d] = val;
}

// (B,MB,80) row-major -> d_out (B,80,MB)
__global__ __launch_bounds__(256)
void k_melT(const float* __restrict__ src, float* __restrict__ dst)
{
  const int idx = blockIdx.x * 256 + threadIdx.x;
  if (idx >= B_ * NM_ * MB_) return;
  const int b = idx / (NM_ * MB_);
  const int r = idx % (NM_ * MB_);
  const int n = r / MB_;
  const int t = r % MB_;
  dst[idx] = src[(size_t)b * MB_ * NM_ + (size_t)t * NM_ + n];
}

// ---------------------------------------------------------------------------
// Orchestration
// ---------------------------------------------------------------------------
static inline void gemm_go(const float* X, const float* W, const float* bia, float* Y,
                           int M, int N, int K, int act, hipStream_t s)
{
  dim3 g((N + 63) / 64, M / 128);
  k_gemm<<<g, 256, 0, s>>>(X, W, bia, Y, M, N, K, act);
}

extern "C" void kernel_launch(void* const* d_in, const int* in_sizes, int n_in,
                              void* d_out, int out_size, void* d_ws, size_t ws_size,
                              hipStream_t stream)
{
  (void)in_sizes; (void)n_in; (void)out_size; (void)ws_size;

  const float* te        = (const float*)d_in[0];
  const int*   lens      = (const int*)  d_in[1];
  const float* e_qkv_w   = (const float*)d_in[2];
  const float* e_qkv_b   = (const float*)d_in[3];
  const float* e_out_w   = (const float*)d_in[4];
  const float* e_out_b   = (const float*)d_in[5];
  const float* e_ln1_w   = (const float*)d_in[6];
  const float* e_ln1_b   = (const float*)d_in[7];
  const float* e_ff1_w   = (const float*)d_in[8];
  const float* e_ff1_b   = (const float*)d_in[9];
  const float* e_ff2_w   = (const float*)d_in[10];
  const float* e_ff2_b   = (const float*)d_in[11];
  const float* e_ln2_w   = (const float*)d_in[12];
  const float* e_ln2_b   = (const float*)d_in[13];
  const float* vp_c1_w   = (const float*)d_in[14];
  const float* vp_c1_b   = (const float*)d_in[15];
  const float* vp_l1_w   = (const float*)d_in[16];
  const float* vp_l1_b   = (const float*)d_in[17];
  const float* vp_c2_w   = (const float*)d_in[18];
  const float* vp_c2_b   = (const float*)d_in[19];
  const float* vp_l2_w   = (const float*)d_in[20];
  const float* vp_l2_b   = (const float*)d_in[21];
  const float* vp_lin_w  = (const float*)d_in[22];
  const float* vp_lin_b  = (const float*)d_in[23];
  const float* d_sa_qkv_w= (const float*)d_in[24];
  const float* d_sa_qkv_b= (const float*)d_in[25];
  const float* d_sa_out_w= (const float*)d_in[26];
  const float* d_sa_out_b= (const float*)d_in[27];
  const float* d_ca_qkv_w= (const float*)d_in[28];
  const float* d_ca_qkv_b= (const float*)d_in[29];
  const float* d_ca_out_w= (const float*)d_in[30];
  const float* d_ca_out_b= (const float*)d_in[31];
  const float* d_ln1_w   = (const float*)d_in[32];
  const float* d_ln1_b   = (const float*)d_in[33];
  const float* d_ln2_w   = (const float*)d_in[34];
  const float* d_ln2_b   = (const float*)d_in[35];
  const float* d_ln3_w   = (const float*)d_in[36];
  const float* d_ln3_b   = (const float*)d_in[37];
  const float* d_ff1_w   = (const float*)d_in[38];
  const float* d_ff1_b   = (const float*)d_in[39];
  const float* d_ff2_w   = (const float*)d_in[40];
  const float* d_ff2_b   = (const float*)d_in[41];
  const float* mel_w     = (const float*)d_in[42];
  const float* mel_b     = (const float*)d_in[43];

  float* ws = (float*)d_ws;
  float* dout = (float*)d_out;

  // workspace layout (floats)
  const size_t F_ENC = 0;                      // B*T*D          = 2,097,152
  const size_t F_Y   = F_ENC + 2097152;        // B*MB*D         = 8,388,608
  const size_t F_ATT = F_Y   + 8388608;        // attn out       = 8,388,608
  const size_t F_TMP = F_ATT + 8388608;        // proj/ff2 out   = 8,388,608
  const size_t F_S   = F_TMP + 8388608;        // big scratch    = 33,554,432
  const size_t F_MSC = F_S   + 33554432;       // sums / maxlen / ends

  float* enc = ws + F_ENC;
  float* Y   = ws + F_Y;
  float* ATT = ws + F_ATT;
  float* TMP = ws + F_TMP;
  float* S   = ws + F_S;
  float* sums = ws + F_MSC;
  int*   maxl = (int*)(ws + F_MSC + 32);
  int*   ends = (int*)(ws + F_MSC + 64);

  const int Me = B_ * T_;    // 8192
  const int Md = B_ * MB_;   // 32768

  // x = text_embeddings
  hipMemcpyAsync(enc, te, (size_t)Me * D_ * sizeof(float),
                 hipMemcpyDeviceToDevice, stream);

  // ---------------- encoder ----------------
  for (int i = 0; i < 4; ++i) {
    gemm_go(enc, e_qkv_w + (size_t)i * 768 * 256, e_qkv_b + i * 768, S,
            Me, 768, 256, 0, stream);
    k_attn<<<dim3(T_ / 64, NH_, B_), 128, 0, stream>>>(
        S, S + 256, S + 512, ATT, lens, 1, T_, T_,
        (long long)T_ * 768, 768, (long long)T_ * 768, 768, (long long)T_ * 768, 768);
    gemm_go(ATT, e_out_w + (size_t)i * 65536, e_out_b + i * 256, TMP,
            Me, 256, 256, 0, stream);
    k_ln<<<Me, 256, 0, stream>>>(enc, TMP, e_ln1_w + i * 256, e_ln1_b + i * 256, enc, 1);
    gemm_go(enc, e_ff1_w + (size_t)i * 262144, e_ff1_b + i * 1024, S,
            Me, 1024, 256, 1, stream);
    gemm_go(S, e_ff2_w + (size_t)i * 262144, e_ff2_b + i * 256, TMP,
            Me, 256, 1024, 0, stream);
    k_ln<<<Me, 256, 0, stream>>>(enc, TMP, e_ln2_w + i * 256, e_ln2_b + i * 256, enc, 1);
  }

  // ---------------- variance predictors (duration, pitch, energy) --------
  const size_t MELSZ = (size_t)B_ * NM_ * MB_;
  float* S2 = S + 2097152;
  for (int j = 0; j < 3; ++j) {
    k_conv3<<<B_ * 256, 256, 0, stream>>>(enc, vp_c1_w + (size_t)j * 196608,
                                          vp_c1_b + j * 256, S, 1);
    k_ln<<<B_ * 256, 256, 0, stream>>>(S, S, vp_l1_w + j * 256, vp_l1_b + j * 256, S2, 0);
    k_conv3<<<B_ * 256, 256, 0, stream>>>(S2, vp_c2_w + (size_t)j * 196608,
                                          vp_c2_b + j * 256, S, 0);
    k_ln<<<B_ * 256, 256, 0, stream>>>(S, S, vp_l2_w + j * 256, vp_l2_b + j * 256, S2, 0);
    k_vlin<<<B_, 256, 0, stream>>>(S2, vp_lin_w + j * 256, vp_lin_b + j,
                                   dout + MELSZ + (size_t)j * 8192);
  }

  // ---------------- length regulate ----------------
  const float* dur = dout + MELSZ;   // duration output (B,256)
  k_rowsum<<<B_, 256, 0, stream>>>(dur, sums);
  k_maxlen<<<1, 32, 0, stream>>>(sums, maxl);
  k_scan<<<B_, 1, 0, stream>>>(dur, lens, maxl, ends);
  k_gather<<<dim3(MB_, B_), 256, 0, stream>>>(enc, ends, Y);

  // ---------------- decoder ----------------
  for (int i = 0; i < 4; ++i) {
    // self-attention
    gemm_go(Y, d_sa_qkv_w + (size_t)i * 768 * 256, d_sa_qkv_b + i * 768, S,
            Md, 768, 256, 0, stream);
    k_attn<<<dim3(MB_ / 64, NH_, B_), 128, 0, stream>>>(
        S, S + 256, S + 512, ATT, lens, 0, MB_, MB_,
        (long long)MB_ * 768, 768, (long long)MB_ * 768, 768, (long long)MB_ * 768, 768);
    gemm_go(ATT, d_sa_out_w + (size_t)i * 65536, d_sa_out_b + i * 256, TMP,
            Md, 256, 256, 0, stream);
    k_ln<<<Md, 256, 0, stream>>>(Y, TMP, d_ln1_w + i * 256, d_ln1_b + i * 256, Y, 1);

    // cross-attention: q from y, k/v from enc
    gemm_go(Y, d_ca_qkv_w + (size_t)i * 768 * 256, d_ca_qkv_b + i * 768, S,
            Md, 256, 256, 0, stream);                               // Q  (B,MB,256)
    gemm_go(enc, d_ca_qkv_w + (size_t)i * 768 * 256 + 65536,
            d_ca_qkv_b + i * 768 + 256, S + 8388608,
            Me, 512, 256, 0, stream);                               // KV (B,T,512)
    k_attn<<<dim3(MB_ / 64, NH_, B_), 128, 0, stream>>>(
        S, S + 8388608, S + 8388608 + 256, ATT, lens, 0, MB_, T_,
        (long long)MB_ * 256, 256, (long long)T_ * 512, 512, (long long)T_ * 512, 512);
    gemm_go(ATT, d_ca_out_w + (size_t)i * 65536, d_ca_out_b + i * 256, TMP,
            Md, 256, 256, 0, stream);
    k_ln<<<Md, 256, 0, stream>>>(Y, TMP, d_ln2_w + i * 256, d_ln2_b + i * 256, Y, 1);

    // feed-forward
    gemm_go(Y, d_ff1_w + (size_t)i * 262144, d_ff1_b + i * 1024, S,
            Md, 1024, 256, 1, stream);
    gemm_go(S, d_ff2_w + (size_t)i * 262144, d_ff2_b + i * 256, TMP,
            Md, 256, 1024, 0, stream);
    k_ln<<<Md, 256, 0, stream>>>(Y, TMP, d_ln3_w + i * 256, d_ln3_b + i * 256, Y, 1);
  }

  // ---------------- mel projection + transpose ----------------
  gemm_go(Y, mel_w, mel_b, S, Md, NM_, 256, 0, stream);
  k_melT<<<(B_ * NM_ * MB_ + 255) / 256, 256, 0, stream>>>(S, dout);
}
